// LSTMMODEL_59665685676274
// MI455X (gfx1250) — compile-verified, matrix operand order checked by reading, not run on hardware
//
#include <hip/hip_runtime.h>

// ---------------- Types ----------------
typedef __attribute__((ext_vector_type(16))) __bf16 v16bf;
typedef __attribute__((ext_vector_type(8)))  float  v8f;

union Frag {
    uint4 q[2];
    v16bf v;
};

#define Bsz 128
#define Tsz 1024
#define INsz 129
#define Ksz 256
#define Msz 128
#define G4K 1024          // 4*K
#define KPADX 160         // input K padded to 5 tiles of 32
#define NTILES_G 64       // 1024 / 16 gate column tiles
#define KT_G 13           // (160 + 256) / 32
#define NTILES_V 8        // 128 / 16
#define KT_V 8            // 256 / 32

__device__ __forceinline__ unsigned short f2bf(float f) {
    union { float f; unsigned int u; } v; v.f = f;
    unsigned int u = v.u;
    unsigned int r = u + 0x7FFFu + ((u >> 16) & 1u);   // round-to-nearest-even
    return (unsigned short)(r >> 16);
}

__device__ __forceinline__ float ftanh(float x) {
#if __has_builtin(__builtin_amdgcn_tanhf)
    return __builtin_amdgcn_tanhf(x);          // v_tanh_f32 (gfx1250 TRANS op)
#else
    float e = __expf(2.0f * x);
    return 1.0f - 2.0f / (e + 1.0f);
#endif
}

__device__ __forceinline__ float fsig(float x) {
    return 0.5f * ftanh(0.5f * x) + 0.5f;      // sigmoid via tanh
}

__device__ __forceinline__ v8f wmma_bf16(const Frag& a, const Frag& b, v8f c) {
    return __builtin_amdgcn_wmma_f32_16x16x32_bf16(
        false, a.v, false, b.v, (short)0, c, false, false);
}

// ---------------- Prep kernels: swizzle weights into WMMA fragment order ----
// Fragment layout (matches ISA 16-bit A/B VGPR layout):
//   flat = ((nt*KT + kt)*32 + lane)*16 + e
//   n = nt*16 + (lane&15),  k = kt*32 + 16*(e>>3) + 8*(lane>>4) + (e&7)

__global__ void prep_bias_k(const float* __restrict__ bih,
                            const float* __restrict__ bhh,
                            float* __restrict__ biasg) {
    int i = blockIdx.x * blockDim.x + threadIdx.x;
    if (i < G4K) biasg[i] = bih[i] + bhh[i];
}

__global__ void prep_wg_k(const float* __restrict__ Wih,
                          const float* __restrict__ Whh,
                          unsigned short* __restrict__ Wg) {
    int idx = blockIdx.x * blockDim.x + threadIdx.x;
    if (idx >= NTILES_G * KT_G * 512) return;
    int e    = idx & 15;
    int lane = (idx >> 4) & 31;
    int rest = idx >> 9;
    int kt   = rest % KT_G;
    int nt   = rest / KT_G;
    int n = nt * 16 + (lane & 15);
    int k = kt * 32 + ((e >> 3) << 4) + ((lane >> 4) << 3) + (e & 7);
    float v;
    if (k < KPADX) v = (k < INsz) ? Wih[n * INsz + k] : 0.0f;   // zero pad 129->160
    else           v = Whh[n * Ksz + (k - KPADX)];
    Wg[idx] = f2bf(v);
}

__global__ void prep_v_k(const float* __restrict__ Vw,
                         unsigned short* __restrict__ Vsw) {
    int idx = blockIdx.x * blockDim.x + threadIdx.x;
    if (idx >= NTILES_V * KT_V * 512) return;
    int e    = idx & 15;
    int lane = (idx >> 4) & 31;
    int rest = idx >> 9;
    int kt   = rest & 7;
    int nt   = rest >> 3;
    int n = nt * 16 + (lane & 15);
    int k = kt * 32 + ((e >> 3) << 4) + ((lane >> 4) << 3) + (e & 7);
    Vsw[idx] = f2bf(Vw[n * Ksz + k]);
}

// ---------------- Persistent LSTM kernel ----------------
// 8 workgroups, each owns a 16-row batch slice for all T steps.
// 512 threads = 16 wave32s. Wave w owns gate-column tiles {w, w+16, w+32, w+48}
// = the SAME 16 columns of i/f/g/o, so the cell update is register-resident.

__global__ __launch_bounds__(512)
void lstm_scan_k(const float* __restrict__ x,            // [B, T, IN]
                 const unsigned short* __restrict__ Wg,  // swizzled [64][13][32][16]
                 const unsigned short* __restrict__ Vsw, // swizzled [8][8][32][16]
                 const float* __restrict__ biasg,        // [1024]
                 const float* __restrict__ Vb,           // [128]
                 float* __restrict__ out) {              // [B, T, M]
    __shared__ __align__(16) unsigned short Ax[16 * KPADX];    // bf16 x_t, 5 KB
    __shared__ __align__(16) unsigned short Hh[2][16 * Ksz];   // bf16 h ping-pong, 2x8 KB
    __shared__ __align__(16) unsigned short Cb[16 * Ksz];      // bf16 cell state, 8 KB

    const int bs   = blockIdx.x;       // batch slice 0..7
    const int tid  = threadIdx.x;
    const int lane = tid & 31;
    const int wave = tid >> 5;
    const int lrow = lane & 15;        // WMMA row/col-in-tile index
    const int lhi  = lane >> 4;        // half-select

    // h0 = 0, c0 = 0, x pad = 0
    for (int i = tid; i < 16 * KPADX; i += 512) Ax[i] = 0;
    for (int i = tid; i < 2 * 16 * Ksz; i += 512) Hh[0][i] = 0;
    __syncthreads();

    // this wave's gate column (same for i/f/g/o tiles)
    const int col = wave * 16 + lrow;
    const float bias0 = biasg[col];            // i : tile nt = wave
    const float bias1 = biasg[256 + col];      // f : tile nt = wave+16
    const float bias2 = biasg[512 + col];      // g : tile nt = wave+32
    const float bias3 = biasg[768 + col];      // o : tile nt = wave+48
    const float vb_n  = Vb[(wave & 7) * 16 + lrow];

    // per-lane weight base pointers (provenance kept clean -> global_load)
    const uint4* wgp = (const uint4*)Wg  + ((size_t)wave * KT_G * 32 + lane) * 2;
    const uint4* vp  = (const uint4*)Vsw + ((size_t)(wave & 7) * KT_V * 32 + lane) * 2;
    // fragment (gate j, kt) at wgp[j*13312 + kt*64], j in {0..3} -> nt = wave+16j

    // x staging slots: row/col fixed across t; running global pointers
    const float* xp[5];
    int xlds[5];
    bool xok[5];
#pragma unroll
    for (int j = 0; j < 5; ++j) {
        int e = tid + j * 512;
        xok[j] = (e < 16 * INsz);
        int r = e / INsz, ci = e - r * INsz;
        if (!xok[j]) { r = 0; ci = 0; }
        xlds[j] = r * KPADX + ci;
        xp[j]   = x + (size_t)(bs * 16 + r) * Tsz * INsz + ci;
    }

    // LDS write bases for h / c (column fixed per thread)
    unsigned short* hb0 = &Hh[1][(8 * lhi) * Ksz + col];   // written when t even
    unsigned short* hb1 = &Hh[0][(8 * lhi) * Ksz + col];   // written when t odd
    unsigned short* cb  = &Cb[(8 * lhi) * Ksz + col];

    // running output pointer (waves 0..7)
    float* op = out + (size_t)(bs * 16 + 8 * lhi) * Tsz * Msz + (wave & 7) * 16 + lrow;

    v8f creg;
#pragma unroll
    for (int r = 0; r < 8; ++r) creg[r] = 0.0f;

    for (int t = 0; t < Tsz; ++t) {
        // Opaque zero offsets: redefined by asm every iteration so the weight
        // fragment loads are NOT loop-invariant (no LICM -> no scratch spill),
        // while the base pointers keep their kernel-arg provenance so the
        // compiler still infers GLOBAL address space (global_load_b128, not
        // flat_load_b128 -> no DScnt coupling with the LDS A-fragment loads).
        unsigned int wz = 0, vz = 0;
        asm volatile("" : "+v"(wz), "+v"(vz));
        const uint4* wgt = wgp + wz;
        const uint4* vpt = vp + vz;

        // ---- stage x_t slice [16 x 129] into Ax (bf16) ----
#pragma unroll
        for (int j = 0; j < 5; ++j) {
            if (xok[j]) {
                Ax[xlds[j]] = f2bf(*xp[j]);
                xp[j] += INsz;
            }
        }
        __syncthreads();

        const unsigned short* hrd = Hh[t & 1];
        unsigned short* hwr = (t & 1) ? hb1 : hb0;

        // ---- gates = [x_t | h] @ [W_ih ; W_hh]^T + (b_ih + b_hh) ----
        v8f acc0, acc1, acc2, acc3;
#pragma unroll
        for (int r = 0; r < 8; ++r) {
            acc0[r] = bias0; acc1[r] = bias1; acc2[r] = bias2; acc3[r] = bias3;
        }

        // software-pipelined over kt: loads for kt+1 issue before WMMAs of kt
        Frag a_cur, b0c, b1c, b2c, b3c;
        {
            const uint4* ap = (const uint4*)(Ax + lrow * KPADX + lhi * 8);
            a_cur.q[0] = ap[0];
            a_cur.q[1] = ap[2];
            b0c.q[0] = wgt[0];         b0c.q[1] = wgt[1];
            b1c.q[0] = wgt[13312];     b1c.q[1] = wgt[13313];
            b2c.q[0] = wgt[2 * 13312]; b2c.q[1] = wgt[2 * 13312 + 1];
            b3c.q[0] = wgt[3 * 13312]; b3c.q[1] = wgt[3 * 13312 + 1];
        }
#pragma unroll
        for (int kt = 0; kt < KT_G; ++kt) {
            Frag a_nxt, b0n, b1n, b2n, b3n;
            if (kt + 1 < KT_G) {
                const int kn = kt + 1;
                b0n.q[0] = wgt[            kn * 64]; b0n.q[1] = wgt[            kn * 64 + 1];
                b1n.q[0] = wgt[13312     + kn * 64]; b1n.q[1] = wgt[13312     + kn * 64 + 1];
                b2n.q[0] = wgt[2 * 13312 + kn * 64]; b2n.q[1] = wgt[2 * 13312 + kn * 64 + 1];
                b3n.q[0] = wgt[3 * 13312 + kn * 64]; b3n.q[1] = wgt[3 * 13312 + kn * 64 + 1];
                const uint4* ap = (kn < 5)
                    ? (const uint4*)(Ax  + lrow * KPADX + lhi * 8 + kn * 32)
                    : (const uint4*)(hrd + lrow * Ksz   + lhi * 8 + (kn - 5) * 32);
                a_nxt.q[0] = ap[0];
                a_nxt.q[1] = ap[2];
            }
            acc0 = wmma_bf16(a_cur, b0c, acc0);
            acc1 = wmma_bf16(a_cur, b1c, acc1);
            acc2 = wmma_bf16(a_cur, b2c, acc2);
            acc3 = wmma_bf16(a_cur, b3c, acc3);
            if (kt + 1 < KT_G) {
                a_cur = a_nxt;
                b0c = b0n; b1c = b1n; b2c = b2n; b3c = b3n;
            }
        }

        // ---- register-resident LSTM cell update ----
#pragma unroll
        for (int r = 0; r < 8; ++r) {
            float si = fsig(acc0[r]);
            float sf = fsig(acc1[r]);
            float tg = ftanh(acc2[r]);
            float so = fsig(acc3[r]);
            float cn = sf * creg[r] + si * tg;
            float hn = so * ftanh(cn);
            creg[r] = cn;
            hwr[r * Ksz] = f2bf(hn);    // h for next step (ping-pong buffer)
            cb[r * Ksz]  = f2bf(cn);    // c for the output projection
        }
        __syncthreads();

        // ---- out = c_new @ V_w^T + V_b  (waves 0..7, one 16-col tile each) ----
        if (wave < 8) {
            v8f oacc;
#pragma unroll
            for (int r = 0; r < 8; ++r) oacc[r] = vb_n;
            // 2-deep pipeline over the 8 V fragments
            Frag vb0, vb1;
            vb0.q[0] = vpt[0];  vb0.q[1] = vpt[1];
            vb1.q[0] = vpt[64]; vb1.q[1] = vpt[65];
#pragma unroll
            for (int kt = 0; kt < KT_V; ++kt) {
                Frag vbn;
                if (kt + 2 < KT_V) {
                    vbn.q[0] = vpt[(kt + 2) * 64];
                    vbn.q[1] = vpt[(kt + 2) * 64 + 1];
                }
                Frag af;
                const uint4* ap = (const uint4*)(Cb + lrow * Ksz + lhi * 8 + kt * 32);
                af.q[0] = ap[0];
                af.q[1] = ap[2];
                oacc = wmma_bf16(af, vb0, oacc);
                vb0 = vb1;
                if (kt + 2 < KT_V) vb1 = vbn;
            }
#pragma unroll
            for (int r = 0; r < 8; ++r)
                op[(size_t)r * Tsz * Msz] = oacc[r];
            op += Msz;
        }
        // next step's post-staging barrier orders the Cb reads above
        // against the next cell update's Cb writes.
    }
}

// ---------------- Launch ----------------
extern "C" void kernel_launch(void* const* d_in, const int* in_sizes, int n_in,
                              void* d_out, int out_size, void* d_ws, size_t ws_size,
                              hipStream_t stream) {
    const float* x   = (const float*)d_in[0];  // [B, T, IN]
    const float* Wih = (const float*)d_in[1];  // [4K, IN]
    const float* Whh = (const float*)d_in[2];  // [4K, K]
    const float* bih = (const float*)d_in[3];  // [4K]
    const float* bhh = (const float*)d_in[4];  // [4K]
    const float* Vw  = (const float*)d_in[5];  // [M, K]
    const float* Vb  = (const float*)d_in[6];  // [M]
    float* out = (float*)d_out;

    // workspace layout
    const size_t wg_elems = (size_t)NTILES_G * KT_G * 512;   // 425984 bf16
    const size_t v_elems  = (size_t)NTILES_V * KT_V * 512;   // 32768 bf16
    unsigned short* Wg  = (unsigned short*)d_ws;
    unsigned short* Vsw = Wg + wg_elems;
    float* biasg        = (float*)(Vsw + v_elems);

    prep_bias_k<<<(G4K + 255) / 256, 256, 0, stream>>>(bih, bhh, biasg);
    prep_wg_k<<<((int)wg_elems + 255) / 256, 256, 0, stream>>>(Wih, Whh, Wg);
    prep_v_k<<<((int)v_elems + 255) / 256, 256, 0, stream>>>(Vw, Vsw);

    lstm_scan_k<<<Bsz / 16, 512, 0, stream>>>(x, Wg, Vsw, biasg, Vb, out);
}
